// DTDRLinear_48498770707315
// MI455X (gfx1250) — matrix operand order repaired and checked by reading, not compiled, not run on hardware
//
#include <hip/hip_runtime.h>
#include <stdint.h>

// ---------------------------------------------------------------------------
// y = (Hadamard(x) @ (Q * s)^T) + bias
//   M = 8192, N = 4096, K = 4096.  Compute-bound f16 WMMA GEMM (~274 GFLOP)
//   vs ~285 MB of HBM traffic, so: FWHT -> f16 once, widen Q -> f16 once
//   (int8 is exact in f16; 32 MB weight stays resident in 192 MB L2),
//   then V_WMMA_F32_16X16X32_F16 with async-to-LDS double buffering.
// Workspace: xh f16 @ 0 (64 MB), Qh f16 @ 64 MB (32 MB)  => needs 96 MB.
// ---------------------------------------------------------------------------

typedef __attribute__((ext_vector_type(16))) _Float16 v16h;
typedef __attribute__((ext_vector_type(8)))  float    v8f;
typedef __attribute__((ext_vector_type(4)))  int      v4i;

#define MDIM 8192
#define NDIM 4096
#define KDIM 4096
#define TK   64       // K-step per staged tile
#define LDSTRIDE 72   // halves per LDS row (64 data + 8 pad => 144B pitch,
                      // bank starts stride 36 mod 64 -> conflict-free B128)

// --- CDNA5 async-copy helpers (ASYNCcnt-tracked; see CDNA5 ISA) ---------
__device__ __forceinline__ void async_ld16(unsigned lds_byte_off, const void* g) {
    asm volatile("global_load_async_to_lds_b128 %0, %1, off"
                 :: "v"(lds_byte_off),
                    "v"((unsigned long long)(uintptr_t)g)
                 : "memory");
}
__device__ __forceinline__ void wait_async0() {
    asm volatile("s_wait_asynccnt 0x0" ::: "memory");
}

// ---------------------------------------------------------------------------
// Kernel 1: FWHT of one 4096-row per block, *1/64, f16 out.
// ---------------------------------------------------------------------------
__global__ void __launch_bounds__(256)
had4096_kernel(const float* __restrict__ x, _Float16* __restrict__ xh)
{
    __shared__ float smem[4096];
    const int tid = threadIdx.x;
    const size_t row = blockIdx.x;
    const float* xr = x + row * (size_t)KDIM;

    float v[16];
    const float4* x4 = (const float4*)(xr + tid * 16);
#pragma unroll
    for (int i = 0; i < 4; ++i) {
        float4 t = x4[i];
        v[i * 4 + 0] = t.x; v[i * 4 + 1] = t.y;
        v[i * 4 + 2] = t.z; v[i * 4 + 3] = t.w;
    }
#pragma unroll
    for (int h = 1; h < 16; h <<= 1) {
#pragma unroll
        for (int i = 0; i < 16; ++i) {
            if ((i & h) == 0) {
                float a = v[i], b = v[i + h];
                v[i] = a + b; v[i + h] = a - b;
            }
        }
    }
#pragma unroll
    for (int i = 0; i < 16; ++i) smem[tid * 16 + i] = v[i];
    __syncthreads();

    for (int h = 16; h < 4096; h <<= 1) {
#pragma unroll
        for (int j = 0; j < 8; ++j) {
            int p = tid + j * 256;
            int i = ((p & ~(h - 1)) << 1) | (p & (h - 1));
            float a = smem[i], b = smem[i + h];
            smem[i] = a + b; smem[i + h] = a - b;
        }
        __syncthreads();
    }

    alignas(16) _Float16 o[16];
#pragma unroll
    for (int i = 0; i < 16; ++i)
        o[i] = (_Float16)(smem[tid * 16 + i] * 0.015625f);
    float4* dst = (float4*)(xh + row * (size_t)KDIM + tid * 16);
    dst[0] = *(const float4*)&o[0];
    dst[1] = *(const float4*)&o[8];
}

// ---------------------------------------------------------------------------
// Kernel 2: one-off int8 -> f16 widening of Q (exact).  ~48 MB traffic.
// ---------------------------------------------------------------------------
__global__ void __launch_bounds__(256)
qwiden_kernel(const int8_t* __restrict__ Q, _Float16* __restrict__ Qh)
{
    size_t base = ((size_t)blockIdx.x * 256 + threadIdx.x) * 16;
    union { v4i v; int8_t c[16]; } qb;
    qb.v = *(const v4i*)(Q + base);
    union { float4 f[2]; _Float16 h[16]; } hb;
#pragma unroll
    for (int i = 0; i < 16; ++i) hb.h[i] = (_Float16)qb.c[i];
    float4* dst = (float4*)(Qh + base);
    dst[0] = hb.f[0];
    dst[1] = hb.f[1];
}

// ---------------------------------------------------------------------------
// Kernel 3: WMMA GEMM.  Block tile 128(M) x 256(N), K-step 64, 256 threads
// = 8 waves in a 2x4 grid; each wave owns 64x64 = 4x4 WMMA tiles
// (32 v_wmma per staged tile, one barrier + one asynccnt wait per tile).
// Double-buffered LDS filled by GLOBAL_LOAD_ASYNC_TO_LDS_B128.
// ---------------------------------------------------------------------------
__global__ void __launch_bounds__(256)
wmma_gemm_kernel(const _Float16* __restrict__ A,     // (M,K) f16
                 const _Float16* __restrict__ Bw,    // (N,K) f16 (widened Q)
                 const float*    __restrict__ s,
                 const float*    __restrict__ bias,
                 float*          __restrict__ C)     // (M,N) f32
{
    __shared__ _Float16 As[2][128 * LDSTRIDE];   // 36 KB
    __shared__ _Float16 Bs[2][256 * LDSTRIDE];   // 72 KB

    const int tid  = threadIdx.x;
    const int lane = tid & 31;
    const int w    = tid >> 5;
    const int wm   = w >> 2;          // 0..1  -> 64-row strip
    const int wn   = w & 3;           // 0..3  -> 64-col strip
    const int m0   = blockIdx.y * 128;
    const int n0   = blockIdx.x * 256;
    const int mrow = lane & 15;
    const int hi   = lane >> 4;       // K-half select per ISA operand layout

    // Staging coords: A -> (row=tid/2, 64B chunk=tid&1); B -> row=tid, 128B.
    const int ar = tid >> 1, ac = tid & 1;
    const _Float16* Ag = A  + (size_t)(m0 + ar) * KDIM + ac * 32;
    const _Float16* Bg = Bw + (size_t)(n0 + tid) * KDIM;

    unsigned asoff[2], bsoff[2];
#pragma unroll
    for (int b = 0; b < 2; ++b) {
        asoff[b] = (unsigned)(uintptr_t)&As[b][ar * LDSTRIDE + ac * 32];
        bsoff[b] = (unsigned)(uintptr_t)&Bs[b][tid * LDSTRIDE];
    }

    auto stage = [&](int buf, int k0) {
#pragma unroll
        for (int c = 0; c < 4; ++c)                       // A: 4 x 16B
            async_ld16(asoff[buf] + c * 16, Ag + k0 + c * 8);
#pragma unroll
        for (int c = 0; c < 8; ++c)                       // B: 8 x 16B
            async_ld16(bsoff[buf] + c * 16, Bg + k0 + c * 8);
    };

    v8f acc[4][4];
#pragma unroll
    for (int mt = 0; mt < 4; ++mt)
#pragma unroll
        for (int nt = 0; nt < 4; ++nt) {
            v8f z = {};
            acc[mt][nt] = z;
        }

    stage(0, 0);
    wait_async0();
    __syncthreads();

    const int KT = KDIM / TK;   // 64 staged tiles
    for (int kt = 0; kt < KT; ++kt) {
        const int cur = kt & 1;
        if (kt + 1 < KT) stage(cur ^ 1, (kt + 1) * TK);

#pragma unroll
        for (int ks = 0; ks < 2; ++ks) {   // two 32-deep WMMA sub-steps
            // A 16x32: VGPR0-3 = K(hi*8+0..7), VGPR4-7 = K(hi*8+16..23)
            v16h afrag[4];
#pragma unroll
            for (int mt = 0; mt < 4; ++mt) {
                int r = wm * 64 + mt * 16 + mrow;
                union { float4 f[2]; v16h v; } u;
                u.f[0] = *(const float4*)&As[cur][r * LDSTRIDE + ks * 32 + hi * 8];
                u.f[1] = *(const float4*)&As[cur][r * LDSTRIDE + ks * 32 + hi * 8 + 16];
                afrag[mt] = u.v;
            }
            // B 32x16: VGPR0-3 = K(hi*8+0..7), VGPR4-7 = K(hi*8+16..23)
            v16h bfrag[4];
#pragma unroll
            for (int nt = 0; nt < 4; ++nt) {
                int c = wn * 64 + nt * 16 + mrow;
                union { float4 f[2]; v16h v; } u;
                u.f[0] = *(const float4*)&Bs[cur][c * LDSTRIDE + ks * 32 + hi * 8];
                u.f[1] = *(const float4*)&Bs[cur][c * LDSTRIDE + ks * 32 + hi * 8 + 16];
                bfrag[nt] = u.v;
            }

#pragma unroll
            for (int mt = 0; mt < 4; ++mt)
#pragma unroll
                for (int nt = 0; nt < 4; ++nt)
                    acc[mt][nt] = __builtin_amdgcn_wmma_f32_16x16x32_f16(
                        false, afrag[mt], false, bfrag[nt],
                        (short)0, acc[mt][nt], false, false);
        }

        wait_async0();      // our fills of the next buffer are done
        __syncthreads();    // everyone's fills done / reads retired
    }

    // Epilogue: y = acc * s[n] + bias[n]
#pragma unroll
    for (int nt = 0; nt < 4; ++nt) {
        int n = n0 + wn * 64 + nt * 16 + mrow;
        float sv = s[n];
        float bv = bias[n];
#pragma unroll
        for (int mt = 0; mt < 4; ++mt) {
            int mbase = m0 + wm * 64 + mt * 16 + hi * 8;
#pragma unroll
            for (int r = 0; r < 8; ++r) {
                C[(size_t)(mbase + r) * NDIM + n] = acc[mt][nt][r] * sv + bv;
            }
        }
    }
}

// ---------------------------------------------------------------------------
extern "C" void kernel_launch(void* const* d_in, const int* in_sizes, int n_in,
                              void* d_out, int out_size, void* d_ws, size_t ws_size,
                              hipStream_t stream) {
    (void)in_sizes; (void)n_in; (void)out_size; (void)ws_size;
    const float*   x    = (const float*)d_in[0];
    const int8_t*  Q    = (const int8_t*)d_in[1];
    const float*   s    = (const float*)d_in[2];
    const float*   bias = (const float*)d_in[3];
    float*         out  = (float*)d_out;

    _Float16* xh = (_Float16*)d_ws;                                     // 64 MB
    _Float16* qh = (_Float16*)((char*)d_ws + (size_t)64 * 1024 * 1024); // 32 MB

    had4096_kernel<<<MDIM, 256, 0, stream>>>(x, xh);

    qwiden_kernel<<<(NDIM * KDIM) / (256 * 16), 256, 0, stream>>>(Q, qh);

    dim3 grid(NDIM / 256, MDIM / 128);
    wmma_gemm_kernel<<<grid, 256, 0, stream>>>(xh, qh, s, bias, out);
}
